// VectorQuantizer_39573828666280
// MI455X (gfx1250) — compile-verified, matrix operand order checked by reading, not run on hardware
//
#include <hip/hip_runtime.h>
#include <hip/hip_bf16.h>

typedef __bf16 v16bf __attribute__((ext_vector_type(16)));
typedef float  v8f   __attribute__((ext_vector_type(8)));

#define N_ROWS   131072   // 64 * 2048
#define EMB_DIM  64
#define N_CODES  1024
#define CHUNK    128      // codes staged in LDS per iteration
#define N_CHUNKS (N_CODES / CHUNK)   // 8
#define TILES_PER_CHUNK (CHUNK / 16) // 8

// Main kernel: 1024 blocks x 256 threads. Each block handles 128 rows;
// each wave (wave32) handles a 16-row M tile against all 1024 codes.
__global__ __launch_bounds__(256) void vq_main(const float* __restrict__ x,
                                               const float* __restrict__ w,
                                               float* __restrict__ wsPartial,
                                               float* __restrict__ outQ,
                                               float* __restrict__ outIdx) {
    // LDS: pre-swizzled B fragments (hi/lo bf16) + per-code norm partials
    __shared__ v16bf sBhi[TILES_PER_CHUNK * 2 * 32];  // 16 KB
    __shared__ v16bf sBlo[TILES_PER_CHUNK * 2 * 32];  // 16 KB
    __shared__ float sNp[2 * CHUNK];                  // 1 KB (two partials/code)
    __shared__ float sRed[8];

    const int t    = threadIdx.x;
    const int wave = t >> 5;
    const int lane = t & 31;
    const int m    = lane & 15;      // row-in-tile / col-in-tile selector
    const int half = lane >> 4;      // lane half per ISA A/B layouts

    const int row0 = blockIdx.x * 128 + wave * 16;

    // ---- Load A fragments (16x32 bf16 layout, ISA 7.12.2) once, hi+lo split.
    // lane<16 : M=lane,   K = {s*32+0..7,  s*32+16..23}
    // lane>=16: M=lane-16,K = {s*32+8..15, s*32+24..31}
    v16bf ahi[2], alo[2];
    {
        const float* xrow = x + (size_t)(row0 + m) * EMB_DIM;
        #pragma unroll
        for (int s = 0; s < 2; ++s) {
            const int b0 = s * 32 + half * 8;
            const int b1 = s * 32 + 16 + half * 8;
            float4 p0 = *(const float4*)(xrow + b0);
            float4 p1 = *(const float4*)(xrow + b0 + 4);
            float4 q0 = *(const float4*)(xrow + b1);
            float4 q1 = *(const float4*)(xrow + b1 + 4);
            float vals[16] = {p0.x,p0.y,p0.z,p0.w, p1.x,p1.y,p1.z,p1.w,
                              q0.x,q0.y,q0.z,q0.w, q1.x,q1.y,q1.z,q1.w};
            #pragma unroll
            for (int e = 0; e < 16; ++e) {
                __bf16 h = (__bf16)vals[e];
                ahi[s][e] = h;
                alo[s][e] = (__bf16)(vals[e] - (float)h);
            }
        }
    }

    float bd[8];
    int   bi[8];
    #pragma unroll
    for (int v = 0; v < 8; ++v) { bd[v] = 3.4e38f; bi[v] = 0; }

    for (int chunk = 0; chunk < N_CHUNKS; ++chunk) {
        __syncthreads();   // previous chunk fully consumed
        // ---- Stage 128 codes: convert f32 -> hi/lo bf16 into exact B layout.
        // Thread t: code kc = t>>1 of chunk, dims (t&1)*32 .. +31 (= k-step s).
        {
            const int kc   = t >> 1;
            const int s    = t & 1;
            const int code = chunk * CHUNK + kc;
            const int tau  = kc >> 4;
            const int n    = kc & 15;
            const float* wp = w + (size_t)code * EMB_DIM + s * 32;
            const int fb = (tau * 2 + s) * 32;
            float nrm = 0.0f;
            #pragma unroll
            for (int j = 0; j < 32; ++j) {
                float val = wp[j];
                nrm += val * val;
                __bf16 h = (__bf16)val;
                __bf16 l = (__bf16)(val - (float)h);
                const int grp   = j >> 3;
                const int wlane = n + ((grp & 1) << 4);
                const int e     = (j & 7) + ((grp & 2) << 2);
                sBhi[fb + wlane][e] = h;
                sBlo[fb + wlane][e] = l;
            }
            sNp[t] = nrm;   // t == 2*kc + s
        }
        __syncthreads();

        // ---- 8 code tiles x (2 k-steps x 3 split WMMAs), then argmin update.
        for (int tau = 0; tau < TILES_PER_CHUNK; ++tau) {
            v8f acc = {0.f,0.f,0.f,0.f,0.f,0.f,0.f,0.f};
            #pragma unroll
            for (int s = 0; s < 2; ++s) {
                v16bf bh = sBhi[(tau * 2 + s) * 32 + lane];
                v16bf bl = sBlo[(tau * 2 + s) * 32 + lane];
                acc = __builtin_amdgcn_wmma_f32_16x16x32_bf16(
                        false, ahi[s], false, bh, (short)0, acc, false, false);
                acc = __builtin_amdgcn_wmma_f32_16x16x32_bf16(
                        false, alo[s], false, bh, (short)0, acc, false, false);
                acc = __builtin_amdgcn_wmma_f32_16x16x32_bf16(
                        false, ahi[s], false, bl, (short)0, acc, false, false);
            }
            const int kc  = tau * 16 + m;              // code within chunk (col N)
            const float nrm = sNp[2 * kc] + sNp[2 * kc + 1];
            const int col = chunk * CHUNK + kc;
            #pragma unroll
            for (int v = 0; v < 8; ++v) {              // rows M=v (+8 for upper half)
                float d = nrm - 2.0f * acc[v];
                if (d < bd[v]) { bd[v] = d; bi[v] = col; }
            }
        }
    }

    // ---- 16-lane butterfly per half (xor 1,2,4,8 stay within a half):
    // argmin with first-index tie-break to match jnp.argmin.
    #pragma unroll
    for (int v = 0; v < 8; ++v) {
        #pragma unroll
        for (int msk = 1; msk < 16; msk <<= 1) {
            float od = __shfl_xor(bd[v], msk, 32);
            int   oi = __shfl_xor(bi[v], msk, 32);
            if (od < bd[v] || (od == bd[v] && oi < bi[v])) { bd[v] = od; bi[v] = oi; }
        }
    }

    // ---- Gather winning code rows, write quantized + indices, accumulate loss.
    float lossAcc = 0.0f;
    #pragma unroll
    for (int v = 0; v < 8; ++v) {
        const int row = row0 + v + 8 * half;           // lanes<16: M=v; >=16: M=v+8
        const int idx = bi[v];                         // uniform within the half
        const float4 wq = *(const float4*)(w + (size_t)idx * EMB_DIM + m * 4);
        const float4 xv = *(const float4*)(x + (size_t)row * EMB_DIM + m * 4);
        *(float4*)(outQ + (size_t)row * EMB_DIM + m * 4) = wq;
        float dx = wq.x - xv.x, dy = wq.y - xv.y, dz = wq.z - xv.z, dw = wq.w - xv.w;
        lossAcc += dx*dx + dy*dy + dz*dz + dw*dw;
        if (m == 0) outIdx[row] = (float)idx;
    }

    // ---- Deterministic block loss reduction -> per-WG partial in d_ws.
    #pragma unroll
    for (int msk = 1; msk < 32; msk <<= 1)
        lossAcc += __shfl_xor(lossAcc, msk, 32);
    if (lane == 0) sRed[wave] = lossAcc;
    __syncthreads();
    if (t == 0) {
        float s = 0.0f;
        #pragma unroll
        for (int i = 0; i < 8; ++i) s += sRed[i];
        wsPartial[blockIdx.x] = s;
    }
}

// Fixed-order final reduction: loss = 1.25 * sum / (N*D)
__global__ __launch_bounds__(256) void vq_reduce(const float* __restrict__ wsPartial,
                                                 float* __restrict__ outLoss) {
    __shared__ float s[256];
    const int t = threadIdx.x;
    s[t] = wsPartial[t] + wsPartial[t + 256] + wsPartial[t + 512] + wsPartial[t + 768];
    __syncthreads();
    for (int o = 128; o > 0; o >>= 1) {
        if (t < o) s[t] += s[t + o];
        __syncthreads();
    }
    if (t == 0) outLoss[0] = 1.25f * s[0] / (float)(N_ROWS * EMB_DIM);
}

extern "C" void kernel_launch(void* const* d_in, const int* in_sizes, int n_in,
                              void* d_out, int out_size, void* d_ws, size_t ws_size,
                              hipStream_t stream) {
    const float* x = (const float*)d_in[0];   // [64,2048,64] f32
    const float* w = (const float*)d_in[1];   // [1024,64]    f32
    float* out     = (float*)d_out;           // [loss | quantized_st | indices]
    float* outQ    = out + 1;
    float* outIdx  = out + 1 + (size_t)N_ROWS * EMB_DIM;
    float* ws      = (float*)d_ws;            // 1024 f32 partials

    vq_main  <<<N_ROWS / 128, 256, 0, stream>>>(x, w, ws, outQ, outIdx);
    vq_reduce<<<1,            256, 0, stream>>>(ws, out);
}
